// VectorQuantizer_16226386444566
// MI455X (gfx1250) — compile-verified, matrix operand order checked by reading, not run on hardware
//
#include <hip/hip_runtime.h>
#include <hip/hip_bf16.h>

#define D     512
#define NE    4096
#define NTOK  32768
#define LDB   520   // padded LDS row stride in halves (1040 B => 4-bank skew/lane)

typedef _Float16 half_t;
typedef __attribute__((ext_vector_type(16))) _Float16 v16h;
typedef __attribute__((ext_vector_type(8)))  _Float16 v8h;
typedef __attribute__((ext_vector_type(8)))  float    v8f;
typedef __attribute__((ext_vector_type(4)))  float    v4f;

// ---------------------------------------------------------------------------
// Kernel 1: codebook f32 -> f16, plus ||e||^2 per code. One wave per row.
// ---------------------------------------------------------------------------
__global__ __launch_bounds__(256) void vq_prep_emb(
    const float* __restrict__ emb, half_t* __restrict__ embh, float* __restrict__ e2)
{
    const int wave = threadIdx.x >> 5;
    const int lane = threadIdx.x & 31;
    const int row  = blockIdx.x * 8 + wave;           // 512 blocks * 8 = 4096 rows
    const float* src = emb + (size_t)row * D;
    half_t*      dst = embh + (size_t)row * D;
    float s = 0.f;
    const int j0 = lane * 16;                          // 16 contiguous floats per lane
#pragma unroll
    for (int h = 0; h < 2; ++h) {
        v4f x0 = *(const v4f*)(src + j0 + h * 8);
        v4f x1 = *(const v4f*)(src + j0 + h * 8 + 4);
        v8h o;
#pragma unroll
        for (int j = 0; j < 4; ++j) {
            s += x0[j] * x0[j] + x1[j] * x1[j];
            o[j]     = (half_t)x0[j];
            o[j + 4] = (half_t)x1[j];
        }
        *(v8h*)(dst + j0 + h * 8) = o;
    }
    for (int off = 16; off >= 1; off >>= 1) s += __shfl_xor(s, off, 32);
    if (lane == 0) e2[row] = s;
}

// ---------------------------------------------------------------------------
// Kernel 2: zero the per-batch presence bitmaps (8 batches x 128 words).
// ---------------------------------------------------------------------------
__global__ void vq_zero(unsigned* __restrict__ presence)
{
    int i = threadIdx.x + blockIdx.x * blockDim.x;
    if (i < 8 * 128) presence[i] = 0u;
}

// ---------------------------------------------------------------------------
// Kernel 3: WMMA distance GEMM + running argmin, with double-buffered
// async global->LDS staging of the f16 codebook (ASYNCcnt-tracked DMA path,
// no VGPR round-trip). Block = 8 waves; each wave owns a 16-token M-tile.
// ---------------------------------------------------------------------------
__device__ __forceinline__ void stage_chunk_async(
    const half_t* __restrict__ gsrc,   // start of 16-code chunk (row-major f16)
    unsigned lds_base,                 // LDS byte offset of destination buffer
    int tid)
{
    // 16 codes x 512 halves = 16 KB; 256 threads x 4 x 16 B, padded rows.
#pragma unroll
    for (int j = 0; j < 4; ++j) {
        const int e = tid + 256 * j;                   // 0..1023 16-byte units
        const int r = e >> 6;                          // 64 units per code row
        const int c = e & 63;
        const unsigned loff = lds_base + (unsigned)(r * (LDB * 2) + c * 16);
        const unsigned long long gaddr = (unsigned long long)(gsrc + e * 8);
        asm volatile("global_load_async_to_lds_b128 %0, %1, off"
                     :: "v"(loff), "v"(gaddr) : "memory");
    }
}

__global__ __launch_bounds__(256) void vq_argmin(
    const float*  __restrict__ z,     // [NTOK, D] f32
    const half_t* __restrict__ embh,  // [NE, D] f16
    const float*  __restrict__ e2,    // [NE]
    int*          __restrict__ index) // [NTOK]
{
    __shared__ alignas(16) half_t Bs[2][16 * LDB];     // double-buffered chunks

    const int tid  = threadIdx.x;
    const int wave = tid >> 5;
    const int lane = tid & 31;
    const int lrow = lane & 15;
    const int lo   = (lane < 16) ? 0 : 8;              // K-slice select (ISA A/B layout)
    const int m0   = (blockIdx.x * 8 + wave) * 16;     // 256 blocks * 128 tokens

    const unsigned bs0 = (unsigned)(size_t)(&Bs[0][0]); // LDS byte offset of buf 0
    const unsigned bufstride = (unsigned)(16 * LDB * 2);

    // Kick off chunk 0 into buffer 0 before doing anything else.
    stage_chunk_async(embh, bs0, tid);

    // Preload the wave's A tile (16 tokens x 512 K) as 16 f16 fragments,
    // converting from f32 on the fly; overlaps with the first async stage.
    v16h Af[16];
    {
        const float* zr = z + (size_t)(m0 + lrow) * D;
#pragma unroll
        for (int ks = 0; ks < 16; ++ks) {
            const int kb = ks * 32 + lo;
            v4f x0 = *(const v4f*)(zr + kb);
            v4f x1 = *(const v4f*)(zr + kb + 4);
            v4f x2 = *(const v4f*)(zr + kb + 16);
            v4f x3 = *(const v4f*)(zr + kb + 20);
            v8h a0, a1;
#pragma unroll
            for (int j = 0; j < 4; ++j) {
                a0[j]     = (half_t)x0[j];
                a0[j + 4] = (half_t)x1[j];
                a1[j]     = (half_t)x2[j];
                a1[j + 4] = (half_t)x3[j];
            }
            Af[ks] = __builtin_shufflevector(a0, a1, 0,1,2,3,4,5,6,7,8,9,10,11,12,13,14,15);
        }
    }

    // Running min/argmin: lane holds column N=lrow, rows M = r + (lane<16?0:8).
    float rmin[8];
    int   rarg[8];
#pragma unroll
    for (int r = 0; r < 8; ++r) { rmin[r] = __builtin_inff(); rarg[r] = 0; }

    const int NCHUNK = NE / 16;                        // 256
    for (int ci = 0; ci < NCHUNK; ++ci) {
        const int p = ci & 1;
        // All waves are done READING buffer (1-p) from iteration ci-1.
        __syncthreads();
        if (ci + 1 < NCHUNK) {
            // Prefetch chunk ci+1 into the other buffer, then retire our own
            // chunk-ci loads (async loads complete in order: <=4 outstanding
            // means the oldest 4 -- chunk ci -- have landed).
            stage_chunk_async(embh + (size_t)(ci + 1) * 16 * D,
                              bs0 + (unsigned)(1 - p) * bufstride, tid);
            asm volatile("s_wait_asynccnt 0x4" ::: "memory");
        } else {
            asm volatile("s_wait_asynccnt 0x0" ::: "memory");
        }
        // All waves' chunk-ci data is in buffer p.
        __syncthreads();

        // 16 chained WMMAs: S = A * B^T over K=512 for a 16x16 score tile.
        v8f acc = {};
        const half_t* br = &Bs[p][lrow * LDB];         // column N=lrow == code row
#pragma unroll
        for (int ks = 0; ks < 16; ++ks) {
            const int kb = ks * 32 + lo;
            v8h b0 = *(const v8h*)(br + kb);
            v8h b1 = *(const v8h*)(br + kb + 16);
            v16h Bf = __builtin_shufflevector(b0, b1, 0,1,2,3,4,5,6,7,8,9,10,11,12,13,14,15);
            acc = __builtin_amdgcn_wmma_f32_16x16x32_f16(
                false, Af[ks], false, Bf, (short)0, acc, false, false);
        }

        // d = ||e||^2 - 2 * (z . e); strict '<' keeps the first (lowest) index.
        const int   n0   = ci * 16;
        const float e2v  = e2[n0 + lrow];
        const int   ncol = n0 + lrow;
#pragma unroll
        for (int r = 0; r < 8; ++r) {
            float dist = __builtin_fmaf(-2.f, acc[r], e2v);
            if (dist < rmin[r]) { rmin[r] = dist; rarg[r] = ncol; }
        }
    }

    // Cross-lane argmin within each 16-lane half (xor <= 8 never crosses halves).
#pragma unroll
    for (int r = 0; r < 8; ++r) {
        float v = rmin[r];
        int   a = rarg[r];
        for (int s = 8; s >= 1; s >>= 1) {
            float ov = __shfl_xor(v, s, 32);
            int   oa = __shfl_xor(a, s, 32);
            if (ov < v || (ov == v && oa < a)) { v = ov; a = oa; }
        }
        if (lrow == 0) index[m0 + (lane >> 4) * 8 + r] = a;
    }
}

// ---------------------------------------------------------------------------
// Kernel 4: z_q = emb[index] gather, squared-diff block partials (deterministic),
// index-as-float output, presence bitmap (idempotent atomicOr).
// One wave per token; 8 tokens per block.
// ---------------------------------------------------------------------------
__global__ __launch_bounds__(256) void vq_gather(
    const float* __restrict__ z, const float* __restrict__ emb,
    const int*   __restrict__ index,
    float* __restrict__ zq, float* __restrict__ idx_out,
    unsigned* __restrict__ presence, float* __restrict__ partials)
{
    __shared__ float wsum[8];
    const int wave = threadIdx.x >> 5;
    const int lane = threadIdx.x & 31;
    const int t    = blockIdx.x * 8 + wave;            // 4096 blocks * 8 = 32768
    const int idx  = index[t];

    const v4f* zr = (const v4f*)(z   + (size_t)t   * D);
    const v4f* er = (const v4f*)(emb + (size_t)idx * D);
    v4f*       qr = (v4f*)(zq + (size_t)t * D);

    float s = 0.f;
#pragma unroll
    for (int j = 0; j < 4; ++j) {                      // 128 v4f / 32 lanes, interleaved
        const int e = j * 32 + lane;
        v4f zv = zr[e];
        v4f ev = er[e];
        qr[e] = ev;
        v4f dv = ev - zv;
        s += dv.x * dv.x + dv.y * dv.y + dv.z * dv.z + dv.w * dv.w;
    }
    for (int off = 16; off >= 1; off >>= 1) s += __shfl_xor(s, off, 32);
    if (lane == 0) {
        wsum[wave] = s;
        idx_out[t] = (float)idx;                       // exact: idx < 4096
        const int b = t >> 12;                         // 4096 tokens per batch
        atomicOr(&presence[b * 128 + (idx >> 5)], 1u << (idx & 31));
    }
    __syncthreads();
    if (threadIdx.x == 0) {
        float tot = 0.f;
        for (int w = 0; w < 8; ++w) tot += wsum[w];    // fixed order -> deterministic
        partials[blockIdx.x] = tot;
    }
}

// ---------------------------------------------------------------------------
// Kernel 5: fixed-order final reduction -> loss, diversity.
// ---------------------------------------------------------------------------
__global__ __launch_bounds__(256) void vq_finalize(
    const float* __restrict__ partials,   // [4096]
    const unsigned* __restrict__ presence,// [1024]
    float* __restrict__ out_loss, float* __restrict__ out_div)
{
    __shared__ float sf[256];
    __shared__ int   si[256];
    const int t = threadIdx.x;
    float s = 0.f;
    for (int i = 0; i < 16; ++i) s += partials[t + 256 * i];
    int c = 0;
    for (int i = t; i < 1024; i += 256) c += __popc(presence[i]);
    sf[t] = s; si[t] = c;
    __syncthreads();
    for (int off = 128; off >= 1; off >>= 1) {
        if (t < off) { sf[t] += sf[t + off]; si[t] += si[t + off]; }
        __syncthreads();
    }
    if (t == 0) {
        *out_loss = 1.25f * sf[0] / (float)((long long)NTOK * D); // (1+BETA)*MSE
        *out_div  = (float)si[0] / (float)NTOK;
    }
}

// ---------------------------------------------------------------------------
extern "C" void kernel_launch(void* const* d_in, const int* in_sizes, int n_in,
                              void* d_out, int out_size, void* d_ws, size_t ws_size,
                              hipStream_t stream)
{
    const float* z   = (const float*)d_in[0];          // [8,16,16,16,512]
    const float* emb = (const float*)d_in[1];          // [4096,512]

    char* ws = (char*)d_ws;                            // ~4.4 MB used
    half_t*   embh     = (half_t*)(ws);                // 4,194,304 B
    float*    e2       = (float*)(ws + 4194304);       //    16,384 B
    int*      index    = (int*)(ws + 4210688);         //   131,072 B
    unsigned* presence = (unsigned*)(ws + 4341760);    //     4,096 B
    float*    partials = (float*)(ws + 4345856);       //    16,384 B

    float* zq    = (float*)d_out;                      // 16,777,216 floats
    float* idxo  = zq + (size_t)NTOK * D;              //     32,768
    float* losso = idxo + NTOK;                        //          1
    float* divo  = losso + 1;                          //          1

    vq_prep_emb<<<512, 256, 0, stream>>>(emb, embh, e2);
    vq_zero<<<4, 256, 0, stream>>>(presence);
    vq_argmin<<<256, 256, 0, stream>>>(z, embh, e2, index);
    vq_gather<<<4096, 256, 0, stream>>>(z, emb, index, zq, idxo, presence, partials);
    vq_finalize<<<1, 256, 0, stream>>>(partials, presence, losso, divo);
}
